// BiMamba2Layer_8289286881569
// MI455X (gfx1250) — matrix operand order repaired and checked by reading, not compiled, not run on hardware
//
#include <hip/hip_runtime.h>
#include <cstdint>
#include <cstddef>

// ---------------- problem constants ----------------
#define DM    512          // D_MODEL
#define DSTT  16           // D_STATE
#define DI    1024         // D_INNER
#define NH    16           // NHEADS
#define HD    64           // HEADDIM
#define CDIM  1056         // CONV_DIM = DI + 2*DSTT
#define DIP   2096         // D_IN_PROJ = 2*DI + 2*DSTT + NH
#define BSZ   8
#define LSEQ  4096
#define NTOK  (BSZ * LSEQ) // 32768 tokens
#define EPSF  1e-5f

typedef __attribute__((ext_vector_type(16))) __bf16 v16bf;
typedef __attribute__((ext_vector_type(8)))  float  v8f;
typedef unsigned int u32x4 __attribute__((ext_vector_type(4)));
typedef int          i32x4 __attribute__((ext_vector_type(4)));
typedef int          i32x8 __attribute__((ext_vector_type(8)));

__device__ __forceinline__ unsigned short f2bf(float f) {
  union { float f; unsigned u; } c; c.f = f;
  unsigned u = c.u;
  u += 0x7FFFu + ((u >> 16) & 1u);   // round-to-nearest-even
  return (unsigned short)(u >> 16);
}
__device__ __forceinline__ __bf16 bfbits(unsigned short s) {
  union { unsigned short u; __bf16 b; } c; c.u = s; return c.b;
}
// LDS byte offset of a __shared__ object (generic addr low 32 bits = LDS offset)
__device__ __forceinline__ unsigned lds_off(const void* p) {
  return (unsigned)(uintptr_t)p;
}

// ---------------- fp32 -> bf16 convert ----------------
__global__ void k_f32_to_bf16(const float* __restrict__ src,
                              unsigned short* __restrict__ dst, size_t n) {
  size_t i = (size_t)blockIdx.x * blockDim.x + threadIdx.x;
  if (i < n) dst[i] = f2bf(src[i]);
}

// ---------------- TDM: load one [rows<=64] x [32] bf16 tile into LDS ----------------
// D# per CDNA5 ISA ch.8: group0 = {flags, lds_addr, global_addr, type=2},
// group1 = {mask/data_size, tensor_dim0/1, tile_dim0/1/2, dim0 stride}.
__device__ __forceinline__ void tdm_load_b_tile(const unsigned short* gsrc,
                                                unsigned lds_byte_off,
                                                int rows_rem, int K) {
  uint64_t ga = (uint64_t)(uintptr_t)gsrc;
  u32x4 g0;
  g0[0] = 1u;                                   // count=1 (user descriptor)
  g0[1] = lds_byte_off;                         // lds_addr
  g0[2] = (unsigned)(ga & 0xFFFFFFFFu);         // global_addr[31:0]
  g0[3] = (unsigned)((ga >> 32) & 0x1FFFFFFu)   // global_addr[56:32]
        | (2u << 30);                           // type = 2 ("image")
  unsigned rows = (rows_rem < 0) ? 0u : (unsigned)rows_rem;
  i32x8 g1;
  g1[0] = (int)(1u << 16);                      // wg_mask=0, data_size=1 (2 bytes)
  g1[1] = (int)(32u << 16);                     // tensor_dim0 = 32 (K elems of this tile)
  g1[2] = (int)((rows & 0xFFFFu) << 16);        // tensor_dim1 low16 (zero-fill OOB rows)
  g1[3] = (int)(((rows >> 16) & 0xFFFFu) | (32u << 16)); // tensor_dim1 hi | tile_dim0=32
  g1[4] = 64;                                   // tile_dim1 = 64, tile_dim2 = 0
  g1[5] = K;                                    // tensor_dim0_stride (data_size units)
  g1[6] = 0;                                    // stride hi / dim1 stride
  g1[7] = 0;
  i32x4 g2 = {0, 0, 0, 0};
  i32x4 g3 = {0, 0, 0, 0};
#if defined(__clang_major__) && __clang_major__ >= 23
  i32x8 g4 = {0, 0, 0, 0, 0, 0, 0, 0};
  __builtin_amdgcn_tensor_load_to_lds(g0, g1, g2, g3, g4, 0);
#else
  __builtin_amdgcn_tensor_load_to_lds(g0, g1, g2, g3, 0);
#endif
}

// ---------------- WMMA bf16 NT GEMM: C[M,N] = A[M,K] * B[N,K]^T ----------------
// 64x64 tile per 256-thread block; 8 waves, each computes a 16x32 strip (2 WMMA/step).
// B tiles arrive via the Tensor Data Mover (double-buffered); A via vector loads.
#define TBM 64
#define TBN 64
#define TBK 32

__global__ __launch_bounds__(256) void k_gemm_wmma(
    const unsigned short* __restrict__ A,   // M x K (bf16 bits), row-major
    const unsigned short* __restrict__ Bw,  // N x K (bf16 bits), row-major
    float* __restrict__ C,                  // M x N f32
    int M, int N, int K, int Lflip)         // Lflip>0: flip token index within seqs of len Lflip
{
  __shared__ unsigned short As[2][TBM][TBK + 8];
  __shared__ unsigned short Bs[2][TBN][TBK];   // TDM-filled, packed row-major

  const int tid  = threadIdx.x;
  const int lane = tid & 31;        // wave32
  const int wave = tid >> 5;        // 0..7
  const int wm   = wave & 3;        // M tile 0..3
  const int wn   = wave >> 2;       // N half 0..1

  // A cooperative load: thread -> 8 elems of one row
  const int lr = tid >> 2;          // 0..63
  const int lc = (tid & 3) * 8;     // 0,8,16,24
  int arow = blockIdx.x * TBM + lr;
  if (Lflip) {
    int b = arow / Lflip, t = arow - b * Lflip;
    arow = b * Lflip + (Lflip - 1 - t);
  }
  const size_t aoff = (size_t)arow * K;

  const int nbase    = blockIdx.y * TBN;
  const int rows_rem = N - nbase;
  const int nk       = K / TBK;

  // prologue: TDM for B buffer 0, vector loads for A buffer 0
  if (wave == 0)
    tdm_load_b_tile(Bw + (size_t)nbase * K, lds_off(&Bs[0][0][0]), rows_rem, K);
  {
    ushort4 ra0 = *(const ushort4*)(A + aoff + lc);
    ushort4 ra1 = *(const ushort4*)(A + aoff + lc + 4);
    *(ushort4*)&As[0][lr][lc]     = ra0;
    *(ushort4*)&As[0][lr][lc + 4] = ra1;
  }

  v8f acc0 = {}, acc1 = {};

  for (int kt = 0; kt < nk; ++kt) {
    const int buf  = kt & 1;
    const bool more = (kt + 1 < nk);
    ushort4 na0, na1;
    if (more) {
      const int k1 = (kt + 1) * TBK;
      na0 = *(const ushort4*)(A + aoff + k1 + lc);
      na1 = *(const ushort4*)(A + aoff + k1 + lc + 4);
      if (kt + 2 < nk)
        __builtin_prefetch(A + aoff + k1 + TBK + lc, 0, 1);  // global_prefetch_b8
    }
    if (wave == 0) __builtin_amdgcn_s_wait_tensorcnt(0);     // B[buf] DMA complete
    __syncthreads();                                         // B[buf]/A[buf] visible
    if (more) {
      if (wave == 0)
        tdm_load_b_tile(Bw + (size_t)nbase * K + (size_t)(kt + 1) * TBK,
                        lds_off(&Bs[buf ^ 1][0][0]), rows_rem, K);
      *(ushort4*)&As[buf ^ 1][lr][lc]     = na0;
      *(ushort4*)&As[buf ^ 1][lr][lc + 4] = na1;
    }

    // A fragment (16x32): lane holds row M=lane&15; lanes0-15: K 0..7,16..23; lanes16-31: K 8..15,24..31
    const int mrow  = wm * 16 + (lane & 15);
    const int khalf = (lane >> 4) * 8;
    v16bf af;
#pragma unroll
    for (int e = 0; e < 16; ++e)
      af[e] = bfbits(As[buf][mrow][khalf + e + ((e >= 8) ? 8 : 0)]);
    // B fragments (32x16 each): lane holds col N=lane&15; lanes0-15 K 0..7,16..23; lanes16-31 K 8..15,24..31
    const int nc0 = wn * 32 + (lane & 15);
    const int kb  = (lane >> 4) * 8;
    v16bf bf0, bf1;
#pragma unroll
    for (int e = 0; e < 16; ++e) bf0[e] = bfbits(Bs[buf][nc0][kb + e + ((e >= 8) ? 8 : 0)]);
#pragma unroll
    for (int e = 0; e < 16; ++e) bf1[e] = bfbits(Bs[buf][nc0 + 16][kb + e + ((e >= 8) ? 8 : 0)]);

    acc0 = __builtin_amdgcn_wmma_f32_16x16x32_bf16(false, af, false, bf0,
                                                   (short)0, acc0, false, false);
    acc1 = __builtin_amdgcn_wmma_f32_16x16x32_bf16(false, af, false, bf1,
                                                   (short)0, acc1, false, false);
    __syncthreads();                                         // reads done before refill
  }

  // C/D layout: lane = col, 8 rows per lane (+8 for lanes >= 16)
  const int gm  = blockIdx.x * TBM + wm * 16 + ((lane >> 4) * 8);
  const int gn0 = nbase + wn * 32 + (lane & 15);
  if (gn0 < N) {
#pragma unroll
    for (int r = 0; r < 8; ++r) C[(size_t)(gm + r) * N + gn0] = acc0[r];
  }
  const int gn1 = gn0 + 16;
  if (gn1 < N) {
#pragma unroll
    for (int r = 0; r < 8; ++r) C[(size_t)(gm + r) * N + gn1] = acc1[r];
  }
}

// ---------------- causal depthwise conv (silu) + dt softplus ----------------
__global__ void k_conv_dt(const float* __restrict__ zx,   // [NTOK][DIP]
                          const float* __restrict__ cw,   // [CDIM][4]
                          const float* __restrict__ cb,   // [CDIM]
                          const float* __restrict__ dtb,  // [NH]
                          float* __restrict__ xc,         // [NTOK][CDIM]
                          float* __restrict__ dt,         // [NTOK][NH]
                          int total) {
  int i = blockIdx.x * blockDim.x + threadIdx.x;
  if (i >= total) return;
  const int CW = CDIM + NH;
  int c  = i % CW;
  int bt = i / CW;
  int t  = bt % LSEQ;
  int b  = bt / LSEQ;
  if (c < CDIM) {
    float acc = cb[c];
#pragma unroll
    for (int k = 0; k < 4; ++k) {
      int ts = t + k - 3;
      if (ts >= 0)
        acc += zx[((size_t)(b * LSEQ + ts)) * DIP + (DI + c)] * cw[c * 4 + k];
    }
    xc[(size_t)bt * CDIM + c] = acc / (1.f + __expf(-acc));   // silu
  } else {
    int h = c - CDIM;
    float v = zx[(size_t)bt * DIP + (DI + CDIM + h)] + dtb[h];
    dt[(size_t)bt * NH + h] = (v > 20.f) ? v : log1pf(__expf(v));  // softplus
  }
}

// ---------------- selective scan: one block per (b,h), lane = headdim p ----------------
__global__ __launch_bounds__(64) void k_scan(const float* __restrict__ xc,  // [NTOK][CDIM]
                                             const float* __restrict__ dt,  // [NTOK][NH]
                                             const float* __restrict__ Alog,// [NH]
                                             const float* __restrict__ Dp,  // [NH]
                                             float* __restrict__ y) {       // [NTOK][DI]
  const int b = blockIdx.x / NH;
  const int h = blockIdx.x % NH;
  const int p = threadIdx.x;               // 0..63
  const float Ah = -__expf(Alog[h]);
  const float Dh = Dp[h];
  float s[DSTT];
#pragma unroll
  for (int n = 0; n < DSTT; ++n) s[n] = 0.f;

  const size_t base = (size_t)b * LSEQ;
  for (int t = 0; t < LSEQ; ++t) {
    const size_t row = base + t;
    const float dtv  = dt[row * NH + h];
    const float dA   = __expf(dtv * Ah);
    const float* xr  = xc + row * CDIM;
    const float xv   = xr[h * HD + p];
    const float coef = dtv * xv;
    const float4* Bv = (const float4*)(xr + DI);
    const float4* Cv = (const float4*)(xr + DI + DSTT);
    float accy = 0.f;
#pragma unroll
    for (int q = 0; q < 4; ++q) {
      float4 Bq = Bv[q], Cq = Cv[q];
      s[4 * q + 0] = s[4 * q + 0] * dA + coef * Bq.x; accy += s[4 * q + 0] * Cq.x;
      s[4 * q + 1] = s[4 * q + 1] * dA + coef * Bq.y; accy += s[4 * q + 1] * Cq.y;
      s[4 * q + 2] = s[4 * q + 2] * dA + coef * Bq.z; accy += s[4 * q + 2] * Cq.z;
      s[4 * q + 3] = s[4 * q + 3] * dA + coef * Bq.w; accy += s[4 * q + 3] * Cq.w;
    }
    y[row * DI + h * HD + p] = accy + Dh * xv;
  }
}

// ---------------- gate (silu(z)) + RMSNorm -> bf16 ----------------
__global__ __launch_bounds__(256) void k_gate_rms(const float* __restrict__ y,  // [NTOK][DI]
                                                  const float* __restrict__ zx, // [NTOK][DIP]
                                                  const float* __restrict__ nw, // [DI]
                                                  unsigned short* __restrict__ yg) {
  __shared__ float red[256];
  const int bt = blockIdx.x, tid = threadIdx.x;
  const float* yr = y  + (size_t)bt * DI;
  const float* zr = zx + (size_t)bt * DIP;   // z = first DI columns
  float g[4]; float ss = 0.f;
#pragma unroll
  for (int j = 0; j < 4; ++j) {
    int idx = j * 256 + tid;
    float z = zr[idx];
    float v = yr[idx] * (z / (1.f + __expf(-z)));
    g[j] = v; ss += v * v;
  }
  red[tid] = ss; __syncthreads();
  for (int s = 128; s > 0; s >>= 1) { if (tid < s) red[tid] += red[tid + s]; __syncthreads(); }
  const float scale = rsqrtf(red[0] / (float)DI + EPSF);
#pragma unroll
  for (int j = 0; j < 4; ++j) {
    int idx = j * 256 + tid;
    yg[(size_t)bt * DI + idx] = f2bf(g[j] * scale * nw[idx]);
  }
}

// ---------------- concat fwd + flipped bwd -> bf16 ----------------
__global__ void k_combine(const float* __restrict__ xf, const float* __restrict__ xb,
                          unsigned short* __restrict__ comb, int total) {
  int i = blockIdx.x * blockDim.x + threadIdx.x;
  if (i >= total) return;
  int j  = i % (2 * DM);
  int bt = i / (2 * DM);
  int t  = bt % LSEQ;
  int b  = bt / LSEQ;
  float v = (j < DM) ? xf[(size_t)bt * DM + j]
                     : xb[((size_t)(b * LSEQ + (LSEQ - 1 - t))) * DM + (j - DM)];
  comb[i] = f2bf(v);
}

// ---------------- residual + bias + LayerNorm ----------------
__global__ __launch_bounds__(256) void k_ln(const float* __restrict__ x,
                                            const float* __restrict__ po,
                                            const float* __restrict__ pb,
                                            const float* __restrict__ lw,
                                            const float* __restrict__ lb,
                                            float* __restrict__ out) {
  __shared__ float rs[256], rq[256];
  const int bt = blockIdx.x, tid = threadIdx.x;
  const float* xr = x  + (size_t)bt * DM;
  const float* pr = po + (size_t)bt * DM;
  float v0 = xr[tid]       + pr[tid]       + pb[tid];
  float v1 = xr[tid + 256] + pr[tid + 256] + pb[tid + 256];
  rs[tid] = v0 + v1; rq[tid] = v0 * v0 + v1 * v1;
  __syncthreads();
  for (int s = 128; s > 0; s >>= 1) {
    if (tid < s) { rs[tid] += rs[tid + s]; rq[tid] += rq[tid + s]; }
    __syncthreads();
  }
  const float mu  = rs[0] / (float)DM;
  const float var = rq[0] / (float)DM - mu * mu;
  const float inv = rsqrtf(var + EPSF);
  out[(size_t)bt * DM + tid]       = (v0 - mu) * inv * lw[tid]       + lb[tid];
  out[(size_t)bt * DM + tid + 256] = (v1 - mu) * inv * lw[tid + 256] + lb[tid + 256];
}

// ---------------- host launch ----------------
extern "C" void kernel_launch(void* const* d_in, const int* in_sizes, int n_in,
                              void* d_out, int out_size, void* d_ws, size_t ws_size,
                              hipStream_t stream) {
  (void)in_sizes; (void)n_in; (void)out_size; (void)ws_size;

  const float* x = (const float*)d_in[0];
  const float* in_proj[2] = { (const float*)d_in[1],  (const float*)d_in[9]  };
  const float* conv_w[2]  = { (const float*)d_in[2],  (const float*)d_in[10] };
  const float* conv_b[2]  = { (const float*)d_in[3],  (const float*)d_in[11] };
  const float* dt_bias[2] = { (const float*)d_in[4],  (const float*)d_in[12] };
  const float* A_log[2]   = { (const float*)d_in[5],  (const float*)d_in[13] };
  const float* Dpar[2]    = { (const float*)d_in[6],  (const float*)d_in[14] };
  const float* norm_w[2]  = { (const float*)d_in[7],  (const float*)d_in[15] };
  const float* out_proj[2]= { (const float*)d_in[8],  (const float*)d_in[16] };
  const float* proj_w = (const float*)d_in[17];
  const float* proj_b = (const float*)d_in[18];
  const float* ln_w   = (const float*)d_in[19];
  const float* ln_b   = (const float*)d_in[20];
  float* out = (float*)d_out;

  char* ws = (char*)d_ws;
  size_t off = 0;
  auto alloc = [&](size_t bytes) -> void* {
    void* p = ws + off;
    off = (off + bytes + 255) & ~(size_t)255;
    return p;
  };

  // persistent
  unsigned short* xbf = (unsigned short*)alloc((size_t)NTOK * DM * 2);
  unsigned short* winbf[2], *woutbf[2];
  winbf[0]  = (unsigned short*)alloc((size_t)DIP * DM * 2);
  winbf[1]  = (unsigned short*)alloc((size_t)DIP * DM * 2);
  woutbf[0] = (unsigned short*)alloc((size_t)DM * DI * 2);
  woutbf[1] = (unsigned short*)alloc((size_t)DM * DI * 2);
  unsigned short* wprojbf = (unsigned short*)alloc((size_t)DM * (2 * DM) * 2);
  float* dirout[2];
  dirout[0] = (float*)alloc((size_t)NTOK * DM * 4);
  dirout[1] = (float*)alloc((size_t)NTOK * DM * 4);
  unsigned short* combbf = (unsigned short*)alloc((size_t)NTOK * (2 * DM) * 2);
  float* projout = (float*)alloc((size_t)NTOK * DM * 4);
  // per-direction scratch (reused fwd then bwd)
  float* zx    = (float*)alloc((size_t)NTOK * DIP * 4);
  float* xc    = (float*)alloc((size_t)NTOK * CDIM * 4);
  float* dtbuf = (float*)alloc((size_t)NTOK * NH * 4);
  float* ybuf  = (float*)alloc((size_t)NTOK * DI * 4);
  unsigned short* ygbf = (unsigned short*)alloc((size_t)NTOK * DI * 2);

  auto cvt = [&](const float* s, unsigned short* d, size_t n) {
    k_f32_to_bf16<<<(unsigned)((n + 255) / 256), 256, 0, stream>>>(s, d, n);
  };
  cvt(x, xbf, (size_t)NTOK * DM);
  cvt(in_proj[0],  winbf[0],  (size_t)DIP * DM);
  cvt(in_proj[1],  winbf[1],  (size_t)DIP * DM);
  cvt(out_proj[0], woutbf[0], (size_t)DM * DI);
  cvt(out_proj[1], woutbf[1], (size_t)DM * DI);
  cvt(proj_w, wprojbf, (size_t)DM * (2 * DM));

  for (int d = 0; d < 2; ++d) {
    const int flip = (d == 1) ? LSEQ : 0;
    // in_proj: zx[NTOK][DIP] = x(bf16, maybe flipped) @ in_proj^T
    k_gemm_wmma<<<dim3(NTOK / TBM, (DIP + TBN - 1) / TBN), 256, 0, stream>>>(
        xbf, winbf[d], zx, NTOK, DIP, DM, flip);
    // conv + dt
    {
      int total = NTOK * (CDIM + NH);
      k_conv_dt<<<(total + 255) / 256, 256, 0, stream>>>(
          zx, conv_w[d], conv_b[d], dt_bias[d], xc, dtbuf, total);
    }
    // selective scan
    k_scan<<<BSZ * NH, 64, 0, stream>>>(xc, dtbuf, A_log[d], Dpar[d], ybuf);
    // gate + RMSNorm -> bf16
    k_gate_rms<<<NTOK, 256, 0, stream>>>(ybuf, zx, norm_w[d], ygbf);
    // out_proj: dirout[d][NTOK][DM] = ygated @ out_proj^T
    k_gemm_wmma<<<dim3(NTOK / TBM, DM / TBN), 256, 0, stream>>>(
        ygbf, woutbf[d], dirout[d], NTOK, DM, DI, 0);
  }

  // concat (flip bwd back) -> bf16
  {
    int total = NTOK * 2 * DM;
    k_combine<<<(total + 255) / 256, 256, 0, stream>>>(dirout[0], dirout[1], combbf, total);
  }
  // final projection: projout = comb @ proj_w^T
  k_gemm_wmma<<<dim3(NTOK / TBM, DM / TBN), 256, 0, stream>>>(
      combbf, wprojbf, projout, NTOK, DM, 2 * DM, 0);
  // residual + bias + LayerNorm
  k_ln<<<NTOK, 256, 0, stream>>>(x, projout, proj_b, ln_w, ln_b, out);
}
